// TABlock_16810501996928
// MI455X (gfx1250) — matrix operand order repaired and checked by reading, not compiled, not run on hardware
//
#include <hip/hip_runtime.h>
#include <math.h>

// ---------------------------------------------------------------------------
// Types for CDNA5 WMMA
// ---------------------------------------------------------------------------
typedef __attribute__((ext_vector_type(16))) __bf16 v16bf;
typedef __attribute__((ext_vector_type(8)))  float  v8f;

union AB16 { unsigned u[8]; v16bf v; };

__device__ __forceinline__ unsigned short f2bf(float f) {
    union { float f; unsigned u; } v; v.f = f;
    unsigned u = v.u;
    unsigned r = u + 0x7FFFu + ((u >> 16) & 1u);   // round-to-nearest-even
    return (unsigned short)(r >> 16);
}
__device__ __forceinline__ unsigned pk2(float a, float b) {
    return (unsigned)f2bf(a) | ((unsigned)f2bf(b) << 16);
}
__device__ __forceinline__ float silu(float x) {
    return x / (1.0f + __expf(-x));
}
__device__ __forceinline__ int clampi(int v, int lo, int hi) {
    return v < lo ? lo : (v > hi ? hi : v);
}

// Problem constants
#define NB   4      // B
#define NT   5      // TP+1 frames
#define NC   256    // C
#define NH   64
#define NW   64
#define XSTR 264    // padded bf16 row stride for LDS X tile (u32 stride 132, uint4 stride 33)

// ---------------------------------------------------------------------------
// Kernel A: fused 1x1-conv (bf16 WMMA) + BN + SiLU + 16x16 max/avg pooling.
// One workgroup per (b, t, 16x16 block). 512 threads = 16 waves.
// Wave w owns output channels d in [w*16, w*16+16) (one WMMA M-tile).
// LDS: X tile (256 spatial x 256 ch, bf16, padded) + bn scale/bias.
// fp is never written to HBM; only pooled max/avg (B,5,C,16) are emitted.
// ---------------------------------------------------------------------------
__global__ __launch_bounds__(512)
void conv_pool_kernel(const float* __restrict__ feature,
                      const float* __restrict__ Wc,
                      const float* __restrict__ bnS_g,
                      const float* __restrict__ bnB_g,
                      float* __restrict__ pmax,
                      float* __restrict__ pavg)
{
    extern __shared__ char smem[];
    unsigned short* Xl = (unsigned short*)smem;              // [256 s][XSTR]
    float* bnS = (float*)(smem + 256 * XSTR * 2);
    float* bnB = bnS + NC;

    const int tid = threadIdx.x;
    const int bid = blockIdx.x;
    const int bt  = bid >> 4;            // (b*5 + t), 0..19
    const int blk = bid & 15;
    const int py  = blk >> 2, px = blk & 3;
    const int cell = py * 4 + px;

    if (tid < NC) { bnS[tid] = bnS_g[tid]; bnB[tid] = bnB_g[tid]; }

    // ---- load X block: feature[bt][c][py*16+sy][px*16+sx] -> Xl[s][c] ----
    {
        const int s  = tid & 255, cp = tid >> 8;
        const int sy = s >> 4,    sx = s & 15;
        const float* fbase = feature + (size_t)bt * NC * 4096
                           + (size_t)(py * 16 + sy) * 64 + (px * 16 + sx);
        for (int c = cp; c < NC; c += 2) {
            Xl[s * XSTR + c] = f2bf(fbase[(size_t)c * 4096]);
        }
    }

    // ---- preload A (W_conv rows for this wave) into WMMA register layout ----
    const int wv   = tid >> 5;           // wave id = M-tile id (0..15)
    const int lane = tid & 31;
    const int lrow = lane & 15;
    const int lhi  = lane >> 4;          // hi half-wave
    AB16 aT[8];                          // 8 K-steps of 32
    {
        const int d = wv * 16 + lrow;
        const float* wr = Wc + (size_t)d * NC;
        for (int kk = 0; kk < 8; ++kk) {
            const int kb = kk * 32 + lhi * 8;
            float4 f0 = *(const float4*)(wr + kb);
            float4 f1 = *(const float4*)(wr + kb + 4);
            float4 f2 = *(const float4*)(wr + kb + 16);
            float4 f3 = *(const float4*)(wr + kb + 20);
            aT[kk].u[0] = pk2(f0.x, f0.y); aT[kk].u[1] = pk2(f0.z, f0.w);
            aT[kk].u[2] = pk2(f1.x, f1.y); aT[kk].u[3] = pk2(f1.z, f1.w);
            aT[kk].u[4] = pk2(f2.x, f2.y); aT[kk].u[5] = pk2(f2.z, f2.w);
            aT[kk].u[6] = pk2(f3.x, f3.y); aT[kk].u[7] = pk2(f3.z, f3.w);
        }
    }
    __syncthreads();

    float sc[8], bi[8];
    for (int r = 0; r < 8; ++r) {
        const int d = wv * 16 + r + lhi * 8;
        sc[r] = bnS[d]; bi[r] = bnB[d];
    }

    float rmax[8], rsum[8];
    for (int r = 0; r < 8; ++r) { rmax[r] = -3.4e38f; rsum[r] = 0.0f; }

    const uint4* XlQ = (const uint4*)smem;      // uint4 row stride = 33

    for (int nt = 0; nt < 8; ++nt) {            // 2 N-tiles per iter
        v8f acc0 = {}; v8f acc1 = {};
        const int s0 = (nt * 2)     * 16 + lrow;
        const int s1 = (nt * 2 + 1) * 16 + lrow;
        for (int kk = 0; kk < 8; ++kk) {
            const int o = kk * 4 + lhi;
            AB16 b0, b1;
            uint4 q0 = XlQ[s0 * 33 + o], q1 = XlQ[s0 * 33 + o + 2];
            b0.u[0] = q0.x; b0.u[1] = q0.y; b0.u[2] = q0.z; b0.u[3] = q0.w;
            b0.u[4] = q1.x; b0.u[5] = q1.y; b0.u[6] = q1.z; b0.u[7] = q1.w;
            uint4 q2 = XlQ[s1 * 33 + o], q3 = XlQ[s1 * 33 + o + 2];
            b1.u[0] = q2.x; b1.u[1] = q2.y; b1.u[2] = q2.z; b1.u[3] = q2.w;
            b1.u[4] = q3.x; b1.u[5] = q3.y; b1.u[6] = q3.z; b1.u[7] = q3.w;
            acc0 = __builtin_amdgcn_wmma_f32_16x16x32_bf16(
                       false, aT[kk].v, false, b0.v, (short)0, acc0, false, false);
            acc1 = __builtin_amdgcn_wmma_f32_16x16x32_bf16(
                       false, aT[kk].v, false, b1.v, (short)0, acc1, false, false);
        }
        // BN + SiLU + fold into running pooled max/sum (per-lane partials)
        for (int r = 0; r < 8; ++r) {
            float x0 = acc0[r] * sc[r] + bi[r]; float v0 = silu(x0);
            float x1 = acc1[r] * sc[r] + bi[r]; float v1 = silu(x1);
            rmax[r] = fmaxf(rmax[r], fmaxf(v0, v1));
            rsum[r] += v0 + v1;
        }
    }

    // reduce across the 16 column-lanes of each half-wave
    for (int m = 1; m <= 8; m <<= 1) {
        for (int r = 0; r < 8; ++r) {
            rmax[r] = fmaxf(rmax[r], __shfl_xor(rmax[r], m, 32));
            rsum[r] += __shfl_xor(rsum[r], m, 32);
        }
    }
    if (lrow == 0) {
        for (int r = 0; r < 8; ++r) {
            const int d = wv * 16 + r + lhi * 8;
            const size_t idx = ((size_t)bt * NC + d) * 16 + cell;
            pmax[idx] = rmax[r];
            pavg[idx] = rsum[r] * (1.0f / 256.0f);
        }
    }
}

// ---------------------------------------------------------------------------
// Kernel B1: tokens -> attn_in (silu(W_in@tok + b_in) + tanh embed) -> Q or K.
// One block per (b, token); n<64 -> past token (produces K), n>=64 -> future
// token (produces Q). Weight rows staged through LDS for coalescing.
// ---------------------------------------------------------------------------
__global__ __launch_bounds__(256)
void token_qk_kernel(const float* __restrict__ pmax, const float* __restrict__ pavg,
                     const float* __restrict__ ptc,  const float* __restrict__ ftc,
                     const float* __restrict__ W_tc, const float* __restrict__ b_tc,
                     const float* __restrict__ W_in, const float* __restrict__ b_in,
                     const float* __restrict__ W_q,  const float* __restrict__ b_q,
                     const float* __restrict__ W_k,  const float* __restrict__ b_k,
                     float* __restrict__ Qb, float* __restrict__ Kb)
{
    __shared__ float tokS[512];
    __shared__ float attnS[256];
    __shared__ float tile[256 * 33];

    const int bid = blockIdx.x;
    const int b   = bid >> 7;
    const int n   = bid & 127;
    const int isF = n >> 6;
    const int tok = n & 63;
    const int t   = isF ? 4 : (tok >> 4);
    const int cell = tok & 15;
    const int d   = threadIdx.x;

    const size_t pbase = ((size_t)(b * 5 + t) * NC) * 16 + cell;
    tokS[d]       = pmax[pbase + (size_t)d * 16];
    tokS[256 + d] = pavg[pbase + (size_t)d * 16];
    __syncthreads();

    // attn_in[d] = silu(W_in[d,:] . tok + b_in[d]) + tanh(tc*W_tc[d]+b_tc[d])
    float acc = b_in[d];
    for (int j0 = 0; j0 < 512; j0 += 32) {
        for (int ii = 0; ii < 32; ++ii) {
            const int lin = ii * 256 + d;
            const int dd = lin >> 5, jj = lin & 31;
            tile[dd * 33 + jj] = W_in[(size_t)dd * 512 + j0 + jj];
        }
        __syncthreads();
        for (int jj = 0; jj < 32; ++jj)
            acc += tile[d * 33 + jj] * tokS[j0 + jj];
        __syncthreads();
    }
    const float tc = (isF ? ftc : ptc)[b * 4 + (tok & 3)];   // jnp.tile -> i % 4
    attnS[d] = silu(acc) + tanhf(tc * W_tc[d] + b_tc[d]);
    __syncthreads();

    // Q (future) or K (past)
    const float* Wqk = isF ? W_q : W_k;
    float acc2 = (isF ? b_q : b_k)[d];
    for (int j0 = 0; j0 < 256; j0 += 32) {
        for (int ii = 0; ii < 32; ++ii) {
            const int lin = ii * 256 + d;
            const int dd = lin >> 5, jj = lin & 31;
            tile[dd * 33 + jj] = Wqk[(size_t)dd * 256 + j0 + jj];
        }
        __syncthreads();
        for (int jj = 0; jj < 32; ++jj)
            acc2 += tile[d * 33 + jj] * attnS[j0 + jj];
        __syncthreads();
    }
    float* dst = isF ? Qb : Kb;
    dst[((size_t)b * 64 + tok) * 256 + d] = acc2;
}

// ---------------------------------------------------------------------------
// Kernel B2a: V[b,n,d] = avgpool(fp[-1])[d,cell] - avgpool(fp[t])[d,cell]
// ---------------------------------------------------------------------------
__global__ __launch_bounds__(256)
void v_kernel(const float* __restrict__ pavg, float* __restrict__ Vb)
{
    const int bid = blockIdx.x;
    const int b = bid >> 6, n = bid & 63;
    const int t = n >> 4, cell = n & 15;
    const int d = threadIdx.x;
    const float last = pavg[(((size_t)b * 5 + 4) * NC + d) * 16 + cell];
    const float past = pavg[(((size_t)b * 5 + t) * NC + d) * 16 + cell];
    Vb[((size_t)b * 64 + n) * 256 + d] = last - past;
}

// ---------------------------------------------------------------------------
// Kernel B2: kv[b,c,d] = sum_n K[b,n,c] * V[b,n,d]   (n over 64)
// ---------------------------------------------------------------------------
__global__ __launch_bounds__(256)
void kv_kernel(const float* __restrict__ Kb, const float* __restrict__ Vb,
               float* __restrict__ kvb)
{
    const int bid = blockIdx.x;
    const int b = bid >> 8, c = bid & 255;
    const int d = threadIdx.x;
    float acc = 0.0f;
    for (int n = 0; n < 64; ++n) {
        const float kk = Kb[((size_t)b * 64 + n) * 256 + c];   // scalar broadcast
        acc += kk * Vb[((size_t)b * 64 + n) * 256 + d];
    }
    kvb[((size_t)b * 256 + c) * 256 + d] = acc;
}

// ---------------------------------------------------------------------------
// Kernel B3: attn[b,q,d] = (Q[b,q,:] . kv[b,:,d]) / sqrt(64)
// ---------------------------------------------------------------------------
__global__ __launch_bounds__(256)
void attn_kernel(const float* __restrict__ Qb, const float* __restrict__ kvb,
                 float* __restrict__ attnb)
{
    const int bid = blockIdx.x;
    const int b = bid >> 6, q = bid & 63;
    const int d = threadIdx.x;
    float acc = 0.0f;
    for (int c = 0; c < 256; ++c) {
        const float qq = Qb[((size_t)b * 64 + q) * 256 + c];   // scalar broadcast
        acc += qq * kvb[((size_t)b * 256 + c) * 256 + d];
    }
    attnb[((size_t)b * 64 + q) * 256 + d] = acc * 0.125f;
}

// ---------------------------------------------------------------------------
// Kernel C: bilinear upsample 4x4 -> 64x64 (half-pixel, edge clamp) fused with
// pred = feature[:, -1] + up * p_attn / 64
// ---------------------------------------------------------------------------
__global__ __launch_bounds__(256)
void upsample_add_kernel(const float* __restrict__ feature,
                         const float* __restrict__ attnb,
                         const float* __restrict__ p_attn,
                         float* __restrict__ out)
{
    const int idx = blockIdx.x * 256 + threadIdx.x;
    const int w  = idx & 63;
    const int h  = (idx >> 6) & 63;
    const int c  = (idx >> 12) & 255;
    const int tf = (idx >> 20) & 3;
    const int b  = idx >> 22;

    const float sh = (h + 0.5f) * 0.0625f - 0.5f;
    const float sw = (w + 0.5f) * 0.0625f - 0.5f;
    const int iy = (int)floorf(sh), ix = (int)floorf(sw);
    const float fy = sh - (float)iy, fx = sw - (float)ix;
    const int y0 = clampi(iy, 0, 3),     y1 = clampi(iy + 1, 0, 3);
    const int x0 = clampi(ix, 0, 3),     x1 = clampi(ix + 1, 0, 3);

    const float* ab = attnb + ((size_t)b * 64 + tf * 16) * 256 + c;
    const float a00 = ab[(y0 * 4 + x0) * 256], a01 = ab[(y0 * 4 + x1) * 256];
    const float a10 = ab[(y1 * 4 + x0) * 256], a11 = ab[(y1 * 4 + x1) * 256];
    const float up = (a00 * (1.0f - fx) + a01 * fx) * (1.0f - fy)
                   + (a10 * (1.0f - fx) + a11 * fx) * fy;

    const float base = feature[(((size_t)b * 5 + 4) * NC + c) * 4096 + h * 64 + w];
    out[idx] = base + up * (p_attn[0] * (1.0f / 64.0f));
}

// ---------------------------------------------------------------------------
// Launch
// ---------------------------------------------------------------------------
extern "C" void kernel_launch(void* const* d_in, const int* in_sizes, int n_in,
                              void* d_out, int out_size, void* d_ws, size_t ws_size,
                              hipStream_t stream)
{
    const float* feature  = (const float*)d_in[0];
    const float* ptc      = (const float*)d_in[1];
    const float* ftc      = (const float*)d_in[2];
    const float* W_conv   = (const float*)d_in[3];
    const float* bn_scale = (const float*)d_in[4];
    const float* bn_bias  = (const float*)d_in[5];
    const float* W_tc     = (const float*)d_in[6];
    const float* b_tc     = (const float*)d_in[7];
    const float* W_in     = (const float*)d_in[8];
    const float* b_in     = (const float*)d_in[9];
    const float* W_q      = (const float*)d_in[10];
    const float* b_q      = (const float*)d_in[11];
    const float* W_k      = (const float*)d_in[12];
    const float* b_k      = (const float*)d_in[13];
    const float* p_attn   = (const float*)d_in[14];

    // workspace partition (floats)
    float* ws    = (float*)d_ws;
    float* pmax  = ws;                 // 4*5*256*16 = 81920
    float* pavg  = pmax + 81920;       // 81920
    float* Qb    = pavg + 81920;       // 4*64*256 = 65536
    float* Kb    = Qb + 65536;         // 65536
    float* Vb    = Kb + 65536;         // 65536
    float* kvb   = Vb + 65536;         // 4*256*256 = 262144
    float* attnb = kvb + 262144;       // 65536
    // total 688128 floats = 2.63 MB

    const int smemA = 256 * XSTR * 2 + 2 * NC * (int)sizeof(float);  // 137216 B
    (void)hipFuncSetAttribute((const void*)conv_pool_kernel,
                              hipFuncAttributeMaxDynamicSharedMemorySize, smemA);

    conv_pool_kernel<<<dim3(320), dim3(512), smemA, stream>>>(
        feature, W_conv, bn_scale, bn_bias, pmax, pavg);

    token_qk_kernel<<<dim3(512), dim3(256), 0, stream>>>(
        pmax, pavg, ptc, ftc, W_tc, b_tc, W_in, b_in, W_q, b_q, W_k, b_k, Qb, Kb);

    v_kernel<<<dim3(256), dim3(256), 0, stream>>>(pavg, Vb);

    kv_kernel<<<dim3(1024), dim3(256), 0, stream>>>(Kb, Vb, kvb);

    attn_kernel<<<dim3(256), dim3(256), 0, stream>>>(Qb, kvb, attnb);

    upsample_add_kernel<<<dim3(65536), dim3(256), 0, stream>>>(
        feature, attnb, p_attn, (float*)d_out);
}
